// AutocorrelationAttention_38603166056837
// MI455X (gfx1250) — matrix operand order repaired and checked by reading, not compiled, not run on hardware
//
#include <hip/hip_runtime.h>
#include <hip/hip_bf16.h>
#include <stdint.h>

// ---------------------------------------------------------------------------
// Autocorrelation attention, restructured for CDNA5 (gfx1250) WMMA.
// B=4, S=2048, D=1024, H=16, dh=64.
// mean_corr = wrapped-diagonal sums of per-head Q K^T (replaces FFT path).
// ---------------------------------------------------------------------------

#define S_LEN 2048
#define D_DIM 1024
#define B_SZ  4
#define H_CNT 16
#define DH    64
#define M_ROWS (B_SZ * S_LEN)     // 8192
#define K_TOP 409                  // S//5

typedef uint32_t u32x4 __attribute__((ext_vector_type(4)));
typedef float    f32x4 __attribute__((ext_vector_type(4)));
typedef float    f32x8 __attribute__((ext_vector_type(8)));
typedef __bf16   bf16x16 __attribute__((ext_vector_type(16)));

struct Frag32B { u32x4 lo; u32x4 hi; };

// Load 16 bf16 from two (possibly discontiguous) 16-byte chunks.
__device__ __forceinline__ bf16x16 ld_frag16(const __bf16* p0, const __bf16* p1) {
  Frag32B f{ *(const u32x4*)p0, *(const u32x4*)p1 };
  return __builtin_bit_cast(bf16x16, f);
}

__device__ __forceinline__ f32x8 wmma_bf16(bf16x16 a, bf16x16 b, f32x8 c) {
  return __builtin_amdgcn_wmma_f32_16x16x32_bf16(false, a, false, b, (short)0, c,
                                                 false, false);
}

// ---------------------------------------------------------------------------
// K0: zero a float buffer
// ---------------------------------------------------------------------------
__global__ __launch_bounds__(256) void k_zero(float* __restrict__ p, int n) {
  int i = blockIdx.x * 256 + threadIdx.x;
  if (i < n) p[i] = 0.f;
}

// ---------------------------------------------------------------------------
// K_T: W (1024x1024 f32, row=in-dim d, col=out n) -> WT (n-major, bf16)
// ---------------------------------------------------------------------------
__global__ __launch_bounds__(256) void k_transpose_bf16(const float* __restrict__ W,
                                                        __bf16* __restrict__ WT) {
  __shared__ float t[32][33];
  const int bx = blockIdx.x & 31;      // tile col (n)
  const int by = blockIdx.x >> 5;      // tile row (d)
  const int tx = threadIdx.x & 31;
  const int ty = threadIdx.x >> 5;     // 0..7
#pragma unroll
  for (int j = 0; j < 4; ++j)
    t[ty + j * 8][tx] = W[(size_t)(by * 32 + ty + j * 8) * D_DIM + bx * 32 + tx];
  __syncthreads();
#pragma unroll
  for (int j = 0; j < 4; ++j)
    WT[(size_t)(bx * 32 + ty + j * 8) * D_DIM + by * 32 + tx] =
        (__bf16)t[tx][ty + j * 8];
}

// ---------------------------------------------------------------------------
// K1: Y[8192x1024] = X[8192x1024] @ W + bias via WMMA bf16.
// WT is bf16, n-major (WT[n][k] = W[k][n]). A converted f32->bf16 in-register.
// Register tiling: each wave owns a 16x64 output strip (4 accumulators).
// All 4 B fragments are loaded into a register array before the WMMA quartet
// so the loads clause together and the 4 WMMAs issue back-to-back.
// Grid: 512 M-tiles x 16 N-strips = 8192 waves / 8 per block = 1024 blocks.
// ---------------------------------------------------------------------------
__global__ __launch_bounds__(256) void k_gemm_proj(const float* __restrict__ X,
                                                   const __bf16* __restrict__ WT,
                                                   const float* __restrict__ bias,
                                                   float* __restrict__ Yf,
                                                   __bf16* __restrict__ Yb,
                                                   int store_bf16) {
  const int lane = threadIdx.x & 31;
  const int wave = threadIdx.x >> 5;
  const int tile = blockIdx.x * 8 + wave;          // 0 .. 8191
  const int ns = tile & 15;                        // 16 N-strips of 64
  const int mt = tile >> 4;                        // 512 M-tiles
  const int i0 = mt * 16, n0 = ns * 64;
  const int lm = lane & 15;
  const int ahalf = (lane >> 4) * 8;               // A K-base: 0 | 8
  const int bhalf = (lane >> 4) * 8;               // B K-base: 0 | 8

  const float*  xrow = X  + (size_t)(i0 + lm) * D_DIM;
  const __bf16* wrow = WT + (size_t)(n0 + lm) * D_DIM;

  f32x8 acc[4];
#pragma unroll
  for (int s = 0; s < 4; ++s) acc[s] = f32x8{0.f, 0.f, 0.f, 0.f, 0.f, 0.f, 0.f, 0.f};

  for (int c = 0; c < D_DIM / 32; ++c) {
    const float* ap = xrow + c * 32 + ahalf;
    if (c + 1 < D_DIM / 32)
      __builtin_prefetch(ap + 32, 0, 3);           // WGP-scope global_prefetch_b8
    // Issue all loads first: 4x b128 (A, f32) + 8x b128 (B, bf16) in one clause.
    f32x4 f0 = *(const f32x4*)(ap);
    f32x4 f1 = *(const f32x4*)(ap + 4);
    f32x4 f2 = *(const f32x4*)(ap + 16);
    f32x4 f3 = *(const f32x4*)(ap + 20);
    bf16x16 bm[4];
#pragma unroll
    for (int s = 0; s < 4; ++s) {
      const __bf16* bp = wrow + (size_t)s * 16 * D_DIM + c * 32 + bhalf;
      bm[s] = ld_frag16(bp, bp + 16);
    }
    bf16x16 a;
#pragma unroll
    for (int e = 0; e < 4; ++e) {
      a[e]      = (__bf16)f0[e];
      a[4 + e]  = (__bf16)f1[e];
      a[8 + e]  = (__bf16)f2[e];
      a[12 + e] = (__bf16)f3[e];
    }
#pragma unroll
    for (int s = 0; s < 4; ++s)                    // back-to-back WMMAs, A reused
      acc[s] = wmma_bf16(a, bm[s], acc[s]);
  }

  const int mbase = i0 + ((lane >> 4) << 3);       // row of acc[*][0]
#pragma unroll
  for (int s = 0; s < 4; ++s) {
    const int col = n0 + s * 16 + lm;
    const float bv = bias[col];
    if (store_bf16) {
#pragma unroll
      for (int r = 0; r < 8; ++r)
        Yb[(size_t)(mbase + r) * D_DIM + col] = (__bf16)(acc[s][r] + bv);
    } else {
#pragma unroll
      for (int r = 0; r < 8; ++r)
        Yf[(size_t)(mbase + r) * D_DIM + col] = acc[s][r] + bv;
    }
  }
}

// ---------------------------------------------------------------------------
// K2: mean_corr[b,h,tau] += sum over wrapped diagonals of Q_h K_h^T.
// Grid: 512 blocks = 64 (b,h) x 8 i-chunks of 256 rows. 8 waves/block.
// Each wave iteration: one Q i-tile (A frags loaded once for dh=64) reused
// across a 4-wide K j-strip; B fragments preloaded so WMMAs issue in pairs.
// LDS tau-accumulator per block (ds_add_f32), then global_atomic_add_f32.
// ---------------------------------------------------------------------------
__global__ __launch_bounds__(256) void k_meancorr(const __bf16* __restrict__ Qb,
                                                  const __bf16* __restrict__ Kb,
                                                  float* __restrict__ C) {
  __shared__ float cacc[S_LEN];
  const int bh = blockIdx.x >> 3;                  // 0..63
  const int ic = blockIdx.x & 7;                   // i-chunk
  const int b = bh >> 4, h = bh & 15;
  for (int i = threadIdx.x; i < S_LEN; i += 256) cacc[i] = 0.f;
  __syncthreads();

  const int lane = threadIdx.x & 31;
  const int wave = threadIdx.x >> 5;
  const int lm = lane & 15;
  const int ahalf = (lane >> 4) * 8;
  const int bhalf = (lane >> 4) * 8;
  const int hoff = h * DH;

  for (int t = wave; t < 16 * 32; t += 8) {        // 16 i-tiles x 32 j-strips
    const int ti = t >> 5, tjs = t & 31;
    const int i0 = ic * 256 + ti * 16;
    const int j0s = tjs * 64;
    const __bf16* qrow = Qb + ((size_t)(b * S_LEN + i0 + lm) << 10) + hoff;
    bf16x16 a0 = ld_frag16(qrow + ahalf,      qrow + ahalf + 16);
    bf16x16 a1 = ld_frag16(qrow + 32 + ahalf, qrow + 32 + ahalf + 16);
    const int mbase = i0 + ((lane >> 4) << 3);
#pragma unroll
    for (int jj = 0; jj < 4; ++jj) {               // reuse A across j-strip
      const int j0 = j0s + jj * 16;
      const __bf16* krow = Kb + ((size_t)(b * S_LEN + j0 + lm) << 10) + hoff;
      bf16x16 b0 = ld_frag16(krow + bhalf,      krow + bhalf + 16);
      bf16x16 b1 = ld_frag16(krow + 32 + bhalf, krow + 32 + bhalf + 16);
      f32x8 acc = {0.f, 0.f, 0.f, 0.f, 0.f, 0.f, 0.f, 0.f};
      acc = wmma_bf16(a0, b0, acc);
      acc = wmma_bf16(a1, b1, acc);
      const int nn = j0 + lm;
#pragma unroll
      for (int r = 0; r < 8; ++r) {
        unsigned tau = ((unsigned)(mbase + r - nn)) & (S_LEN - 1);
        atomicAdd(&cacc[tau], acc[r]);             // ds_add_f32
      }
    }
  }
  __syncthreads();
  for (int i = threadIdx.x; i < S_LEN; i += 256)
    atomicAdd(&C[(size_t)bh * S_LEN + i], cacc[i]); // global_atomic_add_f32
}

// ---------------------------------------------------------------------------
// K3: per (b,h): bitonic top-k (409), softmax over top values, weighted V sum.
// 64 blocks x 256 threads.
// ---------------------------------------------------------------------------
__global__ __launch_bounds__(256) void k_topk_attend(const float* __restrict__ C,
                                                     const float* __restrict__ V,
                                                     float* __restrict__ att) {
  __shared__ float sv[S_LEN];
  __shared__ int   si[S_LEN];
  __shared__ float red[256];
  const int bh = blockIdx.x;
  const int b = bh >> 4, h = bh & 15;
  const int tid = threadIdx.x;
  const float scale = 1.0f / 512.0f;               // mean over dh=64 then /sqrt(64)

  for (int i = tid; i < S_LEN; i += 256) {
    sv[i] = C[(size_t)bh * S_LEN + i] * scale;
    si[i] = i;
  }
  __syncthreads();

  // Bitonic sort, descending by value.
  for (int k = 2; k <= S_LEN; k <<= 1) {
    for (int j = k >> 1; j > 0; j >>= 1) {
      for (int i = tid; i < S_LEN; i += 256) {
        int ixj = i ^ j;
        if (ixj > i) {
          bool desc = ((i & k) == 0);
          float a = sv[i], c2 = sv[ixj];
          if (desc ? (a < c2) : (a > c2)) {
            sv[i] = c2; sv[ixj] = a;
            int ti2 = si[i]; si[i] = si[ixj]; si[ixj] = ti2;
          }
        }
      }
      __syncthreads();
    }
  }

  const float maxv = sv[0];
  float p = 0.f;
  for (int i = tid; i < K_TOP; i += 256) p += expf(sv[i] - maxv);
  red[tid] = p;
  __syncthreads();
  for (int s2 = 128; s2 > 0; s2 >>= 1) {
    if (tid < s2) red[tid] += red[tid + s2];
    __syncthreads();
  }
  const float denom = red[0];
  __syncthreads();

  const int d = tid & 63;
  const int g = tid >> 6;                          // 4 slices over top-k
  float s = 0.f;
  for (int i = g; i < K_TOP; i += 4) {
    float w = expf(sv[i] - maxv);
    s += w * V[((size_t)(b * S_LEN + si[i]) << 10) + h * DH + d];
  }
  red[tid] = s;
  __syncthreads();
  if (g == 0)
    att[(b << 10) + h * DH + d] =
        (red[d] + red[64 + d] + red[128 + d] + red[192 + d]) / denom;
}

// ---------------------------------------------------------------------------
// K4a: y[b,n] = att[b,:] @ wo + bo   (4 x 1024 GEMV, f32)
// ---------------------------------------------------------------------------
__global__ __launch_bounds__(256) void k_outproj(const float* __restrict__ att,
                                                 const float* __restrict__ wo,
                                                 const float* __restrict__ bo,
                                                 float* __restrict__ y) {
  const int b = blockIdx.x;
  const int tid = threadIdx.x;
  float a0 = 0.f, a1 = 0.f, a2 = 0.f, a3 = 0.f;
  const float* ab = att + b * D_DIM;
  for (int d = 0; d < D_DIM; ++d) {
    const float a = ab[d];
    const float* wr = wo + (size_t)d * D_DIM + tid;
    a0 = fmaf(a, wr[0],   a0);
    a1 = fmaf(a, wr[256], a1);
    a2 = fmaf(a, wr[512], a2);
    a3 = fmaf(a, wr[768], a3);
  }
  y[b * D_DIM + tid]       = a0 + bo[tid];
  y[b * D_DIM + tid + 256] = a1 + bo[tid + 256];
  y[b * D_DIM + tid + 512] = a2 + bo[tid + 512];
  y[b * D_DIM + tid + 768] = a3 + bo[tid + 768];
}

// ---------------------------------------------------------------------------
// K4b: out[b,s,:] = y[b,:] broadcast over s (fully coalesced writes)
// ---------------------------------------------------------------------------
__global__ __launch_bounds__(256) void k_broadcast(const float* __restrict__ y,
                                                   float* __restrict__ out) {
  const int idx = blockIdx.x * 256 + threadIdx.x;  // (b,s,n), n fastest
  const int n = idx & (D_DIM - 1);
  const int b = idx >> 21;                         // 2048*1024 = 2^21
  out[idx] = y[(b << 10) + n];
}

// ---------------------------------------------------------------------------
extern "C" void kernel_launch(void* const* d_in, const int* in_sizes, int n_in,
                              void* d_out, int out_size, void* d_ws, size_t ws_size,
                              hipStream_t stream) {
  const float* xq = (const float*)d_in[0];
  const float* xk = (const float*)d_in[1];
  const float* xv = (const float*)d_in[2];
  const float* wq = (const float*)d_in[3];
  const float* bq = (const float*)d_in[4];
  const float* wk = (const float*)d_in[5];
  const float* bk = (const float*)d_in[6];
  const float* wv = (const float*)d_in[7];
  const float* bv = (const float*)d_in[8];
  const float* wo = (const float*)d_in[9];
  const float* bo = (const float*)d_in[10];

  char* p = (char*)d_ws;
  auto alloc = [&](size_t bytes) -> char* {
    char* r = p;
    p += (bytes + 255) & ~(size_t)255;
    return r;
  };
  __bf16* Qb  = (__bf16*)alloc((size_t)M_ROWS * D_DIM * 2);   // 16 MB
  __bf16* Kb  = (__bf16*)alloc((size_t)M_ROWS * D_DIM * 2);   // 16 MB
  float*  Vf  = (float*) alloc((size_t)M_ROWS * D_DIM * 4);   // 32 MB
  __bf16* WTq = (__bf16*)alloc((size_t)D_DIM * D_DIM * 2);    //  2 MB
  __bf16* WTk = (__bf16*)alloc((size_t)D_DIM * D_DIM * 2);
  __bf16* WTv = (__bf16*)alloc((size_t)D_DIM * D_DIM * 2);
  float*  Cm  = (float*) alloc((size_t)B_SZ * H_CNT * S_LEN * 4); // 512 KB
  float*  att = (float*) alloc((size_t)B_SZ * D_DIM * 4);
  float*  y   = (float*) alloc((size_t)B_SZ * D_DIM * 4);

  // 1. Transpose + convert weights to bf16 (n-major) for contiguous B-fragments.
  k_transpose_bf16<<<1024, 256, 0, stream>>>(wq, WTq);
  k_transpose_bf16<<<1024, 256, 0, stream>>>(wk, WTk);
  k_transpose_bf16<<<1024, 256, 0, stream>>>(wv, WTv);

  // 2. Q/K/V projections via WMMA. Q,K -> bf16 (feed correlation), V -> f32.
  k_gemm_proj<<<1024, 256, 0, stream>>>(xq, WTq, bq, nullptr, Qb, 1);
  k_gemm_proj<<<1024, 256, 0, stream>>>(xk, WTk, bk, nullptr, Kb, 1);
  k_gemm_proj<<<1024, 256, 0, stream>>>(xv, WTv, bv, Vf, nullptr, 0);

  // 3. mean autocorrelation = wrapped-diagonal sums of per-head Q K^T.
  k_zero<<<(B_SZ * H_CNT * S_LEN + 255) / 256, 256, 0, stream>>>(
      Cm, B_SZ * H_CNT * S_LEN);
  k_meancorr<<<512, 256, 0, stream>>>(Qb, Kb, Cm);

  // 4. top-k delays + softmax + weighted V gather per head.
  k_topk_attend<<<B_SZ * H_CNT, 256, 0, stream>>>(Cm, Vf, att);

  // 5. output projection (rows identical across s) + broadcast.
  k_outproj<<<B_SZ, 256, 0, stream>>>(att, wo, bo, y);
  k_broadcast<<<(B_SZ * S_LEN * D_DIM) / 256, 256, 0, stream>>>(y, (float*)d_out);
}
